// GroupQueryAttention_54786602827819
// MI455X (gfx1250) — matrix operand order repaired and checked by reading, not compiled, not run on hardware
//
#include <hip/hip_runtime.h>
#include <stdint.h>

// ---------------------------------------------------------------------------
// GQA for MI455X (gfx1250): bf16 WMMA everywhere, f32 accumulate.
//   HIDDEN=2048, N_HEAD=32, N_KV=8, HEAD_DIM=64, B=2, S=2048
// Pipeline: cvt(fp32->bf16) -> GEMM Q/K/V -> flash attention -> GEMM out.
// GEMM: 128x128 block tile, 8 waves (4x2), 32x64 wave tile, double-buffered
//       LDS staging via GLOBAL_LOAD_ASYNC_TO_LDS_B128 when available.
// ---------------------------------------------------------------------------

#define HIDDEN   2048
#define N_HEAD   32
#define N_KV     8
#define HEAD_DIM 64
#define BATCH    2
#define SEQ      2048
#define KV_DIM   (N_KV * HEAD_DIM)   // 512
#define SCALE    0.125f              // 1/sqrt(64)

// Padded LDS row lengths (conflict-free bank strides, 16B-aligned rows)
#define APAD 40    // 80B stride  -> 20-bank step, 16 distinct banks
#define KPAD 72    // 144B stride -> 36-bank step, 16 distinct banks

typedef __attribute__((ext_vector_type(16))) __bf16 v16bf;
typedef __attribute__((ext_vector_type(8)))  __bf16 v8bf;
typedef __attribute__((ext_vector_type(8)))  float  v8f;
typedef __attribute__((__vector_size__(16))) int    v4i_t;   // async builtin arg

union V16 { v16bf v; v8bf h[2]; };

#if defined(__has_builtin)
#if __has_builtin(__builtin_amdgcn_global_load_async_to_lds_b128) && \
    __has_builtin(__builtin_amdgcn_s_wait_asynccnt)
#define USE_ASYNC 1
#endif
#endif

__device__ __forceinline__ __bf16 f2bf(float f) {
  uint32_t u = __builtin_bit_cast(uint32_t, f);
  u += 0x7FFFu + ((u >> 16) & 1u);                 // round-to-nearest-even
  uint16_t hs = (uint16_t)(u >> 16);
  return __builtin_bit_cast(__bf16, hs);
}

__device__ __forceinline__ v8f vzero8() {
  v8f z = {0.f, 0.f, 0.f, 0.f, 0.f, 0.f, 0.f, 0.f};
  return z;
}

// ---------------------------------------------------------------------------
// fp32 -> bf16 conversion (grid-stride)
// ---------------------------------------------------------------------------
__global__ void cvt_bf16_kernel(const float* __restrict__ in,
                                __bf16* __restrict__ out, int n) {
  int i = blockIdx.x * blockDim.x + threadIdx.x;
  int stride = gridDim.x * blockDim.x;
  for (; i < n; i += stride) out[i] = f2bf(in[i]);
}

// ---------------------------------------------------------------------------
// GEMM: C[M,N] = A[M,K](bf16) * W[N,K]^T(bf16) + bias[N]
// Block 256 threads = 8 waves; block tile 128x128; wave tile 32x64; BK=32.
// Double-buffered LDS, one barrier per K-stage, async global->LDS staging.
// ---------------------------------------------------------------------------
__device__ __forceinline__ void gemm_stage(const __bf16* __restrict__ A,
                                           const __bf16* __restrict__ W,
                                           int K, int m0, int n0, int kk,
                                           int tid, __bf16 (&As)[128][APAD],
                                           __bf16 (&Ws)[128][APAD]) {
#pragma unroll
  for (int i = 0; i < 2; ++i) {
    int c   = tid + i * 256;        // 0..511 chunk id
    int row = c >> 2;               // 0..127
    int col = (c & 3) * 8;          // 0,8,16,24 (8 bf16 = 16B)
    const __bf16* ga = &A[(size_t)(m0 + row) * K + kk + col];
    const __bf16* gw = &W[(size_t)(n0 + row) * K + kk + col];
#ifdef USE_ASYNC
    __builtin_amdgcn_global_load_async_to_lds_b128((v4i_t*)ga,
                                                   (v4i_t*)&As[row][col], 0, 0);
    __builtin_amdgcn_global_load_async_to_lds_b128((v4i_t*)gw,
                                                   (v4i_t*)&Ws[row][col], 0, 0);
#else
    *(v8bf*)&As[row][col] = *(const v8bf*)ga;
    *(v8bf*)&Ws[row][col] = *(const v8bf*)gw;
#endif
  }
}

__global__ __launch_bounds__(256)
void gemm_bf16_wmma(const __bf16* __restrict__ A, const __bf16* __restrict__ W,
                    const float* __restrict__ bias,
                    __bf16* __restrict__ outB, float* __restrict__ outF,
                    int M, int N, int K, int writeF32) {
  __shared__ alignas(64) __bf16 As[2][128][APAD];
  __shared__ alignas(64) __bf16 Ws[2][128][APAD];

  const int tid  = threadIdx.x;
  const int lane = tid & 31;
  const int wave = tid >> 5;
  const int wr   = wave >> 1;       // 0..3 -> 32-row band
  const int wc   = wave & 1;        // 0..1 -> 64-col band
  const int m0   = blockIdx.y * 128;
  const int n0   = blockIdx.x * 128;

  const int hi    = lane >> 4;
  const int l15   = lane & 15;
  const int koffA = hi * 8;         // A-fragment K offset (ISA layout)
  const int koffB = hi * 16;        // unused; B fragments also load with koffA

  v8f acc[2][4];
#pragma unroll
  for (int i = 0; i < 2; ++i)
#pragma unroll
    for (int j = 0; j < 4; ++j) acc[i][j] = vzero8();

  const int NS = K >> 5;            // K / 32 stages
  gemm_stage(A, W, K, m0, n0, 0, tid, As[0], Ws[0]);

  for (int s = 0; s < NS; ++s) {
    const int buf = s & 1;
#ifdef USE_ASYNC
    __builtin_amdgcn_s_wait_asynccnt(0);   // this wave's stage-s data in LDS
#endif
    __syncthreads();                        // all waves staged + prev compute done
    if (s + 1 < NS)
      gemm_stage(A, W, K, m0, n0, (s + 1) * 32, tid, As[buf ^ 1], Ws[buf ^ 1]);

    // ---- fragments from LDS (conflict-free 80B row stride) ----
    V16 af[2];
#pragma unroll
    for (int i = 0; i < 2; ++i) {
      const __bf16* p = &As[buf][wr * 32 + i * 16 + l15][0];
      af[i].h[0] = *(const v8bf*)&p[koffA];
      af[i].h[1] = *(const v8bf*)&p[16 + koffA];
    }
    V16 bfr[4];
#pragma unroll
    for (int j = 0; j < 4; ++j) {
      const __bf16* p = &Ws[buf][wc * 64 + j * 16 + l15][0];
      bfr[j].h[0] = *(const v8bf*)&p[koffA];
      bfr[j].h[1] = *(const v8bf*)&p[16 + koffA];
    }
#pragma unroll
    for (int i = 0; i < 2; ++i)
#pragma unroll
      for (int j = 0; j < 4; ++j)
        acc[i][j] = __builtin_amdgcn_wmma_f32_16x16x32_bf16(
            false, af[i].v, false, bfr[j].v, (short)0, acc[i][j], false, false);
  }

  // ---- epilogue: C/D layout row r + 8*hi, col = lane&15 in 16-col band ----
#pragma unroll
  for (int i = 0; i < 2; ++i)
#pragma unroll
    for (int j = 0; j < 4; ++j)
#pragma unroll
      for (int r = 0; r < 8; ++r) {
        int row = m0 + wr * 32 + i * 16 + r + hi * 8;
        int col = n0 + wc * 64 + j * 16 + l15;
        float v = acc[i][j][r] + bias[col];
        if (writeF32) outF[(size_t)row * N + col] = v;
        else          outB[(size_t)row * N + col] = f2bf(v);
      }
}

// ---------------------------------------------------------------------------
// Flash attention for GQA.
// Grid: (SEQ/64, BATCH*N_HEAD). Block: 128 threads = 4 waves.
// Each wave owns 16 query rows; key loop in tiles of 32 with online softmax.
// V is transposed at staging time (Vt[d][l]) so P*V B-fragments are clean
// ds_load_b128 pairs instead of 16-element scalar gathers.
// O: [B,H,S,D] bf16 contiguous (matches reference's flat reshape).
// ---------------------------------------------------------------------------
__global__ __launch_bounds__(128)
void flash_gqa_wmma(const __bf16* __restrict__ Q, const __bf16* __restrict__ Kc,
                    const __bf16* __restrict__ Vc, __bf16* __restrict__ O) {
  __shared__ alignas(64) __bf16 Ks[32][KPAD];        // keys x d   (144B rows)
  __shared__ alignas(64) __bf16 Vt[HEAD_DIM][APAD];  // d x keys   (80B rows)
  __shared__ alignas(64) __bf16 Ps[4][16][APAD];     // per-wave P (80B rows)

  const int tid  = threadIdx.x;
  const int lane = tid & 31;
  const int wave = tid >> 5;
  const int bh   = blockIdx.y;
  const int b    = bh >> 5;           // / N_HEAD
  const int h    = bh & 31;
  const int kv   = h >> 2;            // h / (N_HEAD/N_KV)
  const int q0   = blockIdx.x * 64 + wave * 16;

  const int hi    = lane >> 4;
  const int l15   = lane & 15;
  const int koffA = hi * 8;
  const int koffB = hi * 16;

  // ---- load Q fragments for this wave's 16 rows (kept in VGPRs) ----
  const __bf16* qptr =
      Q + ((size_t)(b * SEQ + q0 + l15) * HIDDEN) + h * HEAD_DIM;
  V16 qa0, qa1;
  qa0.h[0] = *(const v8bf*)&qptr[koffA];
  qa0.h[1] = *(const v8bf*)&qptr[16 + koffA];
  qa1.h[0] = *(const v8bf*)&qptr[32 + koffA];
  qa1.h[1] = *(const v8bf*)&qptr[48 + koffA];

  float mrow[8], lrow[8];
#pragma unroll
  for (int r = 0; r < 8; ++r) { mrow[r] = -3.0e38f; lrow[r] = 0.f; }
  v8f acc[4] = {vzero8(), vzero8(), vzero8(), vzero8()};

  for (int k0 = 0; k0 < SEQ; k0 += 32) {
    // ---- stage K (async when possible) and transposed V ----
#pragma unroll
    for (int i = 0; i < 2; ++i) {
      int c   = tid + i * 128;      // 256 chunks of 8 bf16
      int row = c >> 3;             // 0..31 key row
      int col = (c & 7) * 8;        // 0..56 d offset
      size_t gbase =
          (size_t)(b * SEQ + k0 + row) * KV_DIM + kv * HEAD_DIM + col;
#ifdef USE_ASYNC
      __builtin_amdgcn_global_load_async_to_lds_b128(
          (v4i_t*)&Kc[gbase], (v4i_t*)&Ks[row][col], 0, 0);
#else
      *(v8bf*)&Ks[row][col] = *(const v8bf*)&Kc[gbase];
#endif
      v8bf vv = *(const v8bf*)&Vc[gbase];
#pragma unroll
      for (int j = 0; j < 8; ++j) Vt[col + j][row] = vv[j];
    }
#ifdef USE_ASYNC
    __builtin_amdgcn_s_wait_asynccnt(0);
#endif
    __syncthreads();

    // ---- scores: S = Q * K^T  (two key sub-tiles x two K=32 d-steps) ----
    v8f s0 = vzero8();
    v8f s1 = vzero8();
    {
      V16 kb00, kb01, kb10, kb11;
      const __bf16* p0 = &Ks[l15][0];
      const __bf16* p1 = &Ks[16 + l15][0];
      kb00.h[0] = *(const v8bf*)&p0[koffA];      kb00.h[1] = *(const v8bf*)&p0[16 + koffA];
      kb01.h[0] = *(const v8bf*)&p1[koffA];      kb01.h[1] = *(const v8bf*)&p1[16 + koffA];
      kb10.h[0] = *(const v8bf*)&p0[32 + koffA]; kb10.h[1] = *(const v8bf*)&p0[48 + koffA];
      kb11.h[0] = *(const v8bf*)&p1[32 + koffA]; kb11.h[1] = *(const v8bf*)&p1[48 + koffA];
      s0 = __builtin_amdgcn_wmma_f32_16x16x32_bf16(false, qa0.v, false, kb00.v,
                                                   (short)0, s0, false, false);
      s1 = __builtin_amdgcn_wmma_f32_16x16x32_bf16(false, qa0.v, false, kb01.v,
                                                   (short)0, s1, false, false);
      s0 = __builtin_amdgcn_wmma_f32_16x16x32_bf16(false, qa1.v, false, kb10.v,
                                                   (short)0, s0, false, false);
      s1 = __builtin_amdgcn_wmma_f32_16x16x32_bf16(false, qa1.v, false, kb11.v,
                                                   (short)0, s1, false, false);
    }

    // ---- online softmax (row reductions across 16-lane half-waves) ----
#pragma unroll
    for (int r = 0; r < 8; ++r) {
      float sm0 = s0[r] * SCALE;
      float sm1 = s1[r] * SCALE;
      float v = fmaxf(sm0, sm1);
#pragma unroll
      for (int off = 1; off < 16; off <<= 1)
        v = fmaxf(v, __shfl_xor(v, off, 16));
      float nm  = fmaxf(mrow[r], v);
      float fac = __expf(mrow[r] - nm);
      mrow[r] = nm;
      float p0 = __expf(sm0 - nm);
      float p1 = __expf(sm1 - nm);
      float ps = p0 + p1;
#pragma unroll
      for (int off = 1; off < 16; off <<= 1)
        ps += __shfl_xor(ps, off, 16);
      lrow[r] = lrow[r] * fac + ps;
      acc[0][r] *= fac;
      acc[1][r] *= fac;
      acc[2][r] *= fac;
      acc[3][r] *= fac;
      // C-layout -> LDS (row = r + 8*hi, cols = key index)
      Ps[wave][r + hi * 8][l15]      = f2bf(p0);
      Ps[wave][r + hi * 8][16 + l15] = f2bf(p1);
    }

    // ---- O += P * V : A-frag of P, B-frags from transposed V ----
    V16 pa;
    pa.h[0] = *(const v8bf*)&Ps[wave][l15][koffA];
    pa.h[1] = *(const v8bf*)&Ps[wave][l15][16 + koffA];
#pragma unroll
    for (int nb = 0; nb < 4; ++nb) {
      V16 vb;
      const __bf16* p = &Vt[nb * 16 + l15][0];   // lane owns column d
      vb.h[0] = *(const v8bf*)&p[koffA];
      vb.h[1] = *(const v8bf*)&p[16 + koffA];
      acc[nb] = __builtin_amdgcn_wmma_f32_16x16x32_bf16(
          false, pa.v, false, vb.v, (short)0, acc[nb], false, false);
    }
    __syncthreads();
  }

  // ---- normalize and store O as [B, H, S, D] bf16 ----
#pragma unroll
  for (int r = 0; r < 8; ++r) {
    float inv = 1.0f / lrow[r];
    size_t rowbase =
        ((size_t)(b * N_HEAD + h) * SEQ + q0 + r + hi * 8) * HEAD_DIM;
#pragma unroll
    for (int nb = 0; nb < 4; ++nb)
      O[rowbase + nb * 16 + l15] = f2bf(acc[nb][r] * inv);
  }
}

// ---------------------------------------------------------------------------
extern "C" void kernel_launch(void* const* d_in, const int* in_sizes, int n_in,
                              void* d_out, int out_size, void* d_ws,
                              size_t ws_size, hipStream_t stream) {
  (void)in_sizes; (void)n_in; (void)out_size; (void)ws_size;

  const float* X  = (const float*)d_in[0];
  const float* Wq = (const float*)d_in[1];
  const float* bq = (const float*)d_in[2];
  const float* Wk = (const float*)d_in[3];
  const float* bk = (const float*)d_in[4];
  const float* Wv = (const float*)d_in[5];
  const float* bv = (const float*)d_in[6];
  const float* Wo = (const float*)d_in[7];
  const float* bo = (const float*)d_in[8];
  float* out = (float*)d_out;

  char* ws = (char*)d_ws;
  size_t off = 0;
  __bf16* Xb  = (__bf16*)(ws + off); off += (size_t)BATCH * SEQ * HIDDEN * 2;
  __bf16* Wqb = (__bf16*)(ws + off); off += (size_t)HIDDEN * HIDDEN * 2;
  __bf16* Wkb = (__bf16*)(ws + off); off += (size_t)KV_DIM * HIDDEN * 2;
  __bf16* Wvb = (__bf16*)(ws + off); off += (size_t)KV_DIM * HIDDEN * 2;
  __bf16* Wob = (__bf16*)(ws + off); off += (size_t)HIDDEN * HIDDEN * 2;
  __bf16* Qb  = (__bf16*)(ws + off); off += (size_t)BATCH * SEQ * HIDDEN * 2;
  __bf16* Kb  = (__bf16*)(ws + off); off += (size_t)BATCH * SEQ * KV_DIM * 2;
  __bf16* Vb  = (__bf16*)(ws + off); off += (size_t)BATCH * SEQ * KV_DIM * 2;
  __bf16* Ob  = (__bf16*)(ws + off); off += (size_t)BATCH * SEQ * HIDDEN * 2;

  // 1) fp32 -> bf16 conversions
  cvt_bf16_kernel<<<4096, 256, 0, stream>>>(X,  Xb,  BATCH * SEQ * HIDDEN);
  cvt_bf16_kernel<<<2048, 256, 0, stream>>>(Wq, Wqb, HIDDEN * HIDDEN);
  cvt_bf16_kernel<<<1024, 256, 0, stream>>>(Wk, Wkb, KV_DIM * HIDDEN);
  cvt_bf16_kernel<<<1024, 256, 0, stream>>>(Wv, Wvb, KV_DIM * HIDDEN);
  cvt_bf16_kernel<<<2048, 256, 0, stream>>>(Wo, Wob, HIDDEN * HIDDEN);

  const int M = BATCH * SEQ;  // 4096 flat rows

  // 2) projections (bf16 out)
  gemm_bf16_wmma<<<dim3(HIDDEN / 128, M / 128), 256, 0, stream>>>(
      Xb, Wqb, bq, Qb, nullptr, M, HIDDEN, HIDDEN, 0);
  gemm_bf16_wmma<<<dim3(KV_DIM / 128, M / 128), 256, 0, stream>>>(
      Xb, Wkb, bk, Kb, nullptr, M, KV_DIM, HIDDEN, 0);
  gemm_bf16_wmma<<<dim3(KV_DIM / 128, M / 128), 256, 0, stream>>>(
      Xb, Wvb, bv, Vb, nullptr, M, KV_DIM, HIDDEN, 0);

  // 3) flash attention: O laid out [B,H,S,D] == reference's flat reshape
  flash_gqa_wmma<<<dim3(SEQ / 64, BATCH * N_HEAD), 128, 0, stream>>>(
      Qb, Kb, Vb, Ob);

  // 4) output projection (fp32 out straight to d_out)
  gemm_bf16_wmma<<<dim3(HIDDEN / 128, M / 128), 256, 0, stream>>>(
      Ob, Wob, bo, nullptr, out, M, HIDDEN, HIDDEN, 1);
}
